// GraphSAGELinkPredictor_42176578846858
// MI455X (gfx1250) — matrix operand order repaired and checked by reading, not compiled, not run on hardware
//
#include <hip/hip_runtime.h>
#include <hip/hip_bf16.h>

typedef __attribute__((ext_vector_type(16))) _Float16 v16h;
typedef __attribute__((ext_vector_type(8)))  _Float16 v8h;
typedef __attribute__((ext_vector_type(4)))  _Float16 v4h;
typedef __attribute__((ext_vector_type(8)))  float    v8f;

// ---------------------------------------------------------------------------
// Dual GEMM: Cl = X @ Wl^T, Cr = X @ Wr^T  (fp32 memory, f16 MACs, f32 acc).
// Both projections share the A (X) tile -> X is read once, each A-fragment
// feeds 4 independent WMMAs per iteration (2 K-halves x 2 weights).
//
// Block = 128 threads (4 waves, wave32). Block tile = 16 M-rows x 64 N-cols.
// K-step = 64. gridDim.x = M/16, gridDim.y = N/64. K_DIM multiple of 64.
//
// LDS rows padded 64 -> 72 halves (144 B = 9*16 B): chunks stay 16B-aligned,
// 36-dword row stride => period-16 bank pattern, conflict-free frag reads.
//
// Fragment layouts used here (contiguous K order, same for A and B):
//   A (16x32 f16): lane l holds row M=l&15; elems 0..15 = K 16*hi+0..15
//                  (hi = l>>4).
//   B (32x16 f16): lane l holds col N=l&15; elems 0..15 = K 16*hi+0..15.
//   C (16x16 f32): VGPR r -> M = r + 8*hi, N = l&15.
// ---------------------------------------------------------------------------
template <int K_DIM>
__global__ __launch_bounds__(128) void sage_gemm_dual(
    const float* __restrict__ X,
    const float* __restrict__ Wl, const float* __restrict__ Wr,
    float* __restrict__ Cl, float* __restrict__ Cr, int M, int N) {
  constexpr int LDS_STRIDE = 72;  // halves per row (64 data + 8 pad)
  __shared__ __align__(16) _Float16 sA [16][LDS_STRIDE];
  __shared__ __align__(16) _Float16 sBl[64][LDS_STRIDE];
  __shared__ __align__(16) _Float16 sBr[64][LDS_STRIDE];

  const int tid  = threadIdx.x;
  const int wave = tid >> 5;
  const int lane = tid & 31;
  const int m0   = blockIdx.x * 16;
  const int n0   = blockIdx.y * 64;

  const int mrow = lane & 15;
  const int hi   = lane >> 4;
  const int ncol = wave * 16 + (lane & 15);

  // Staging: chunk idx -> row = idx>>4, col = (idx&15)*4.
  float4 aReg[2];   // sA : 16x64 = 256 float4-chunks / 128 thr = 2 each
  float4 blReg[8];  // sBl: 64x64 = 1024 chunks / 128 thr = 8 each
  float4 brReg[8];  // sBr: same

  auto loadRegs = [&](int k0) {
#pragma unroll
    for (int i = 0; i < 2; ++i) {
      int idx = i * 128 + tid;
      int r = idx >> 4, c = (idx & 15) * 4;
      aReg[i] = *(const float4*)&X[(size_t)(m0 + r) * K_DIM + (k0 + c)];
    }
#pragma unroll
    for (int i = 0; i < 8; ++i) {
      int idx = i * 128 + tid;
      int r = idx >> 4, c = (idx & 15) * 4;
      blReg[i] = *(const float4*)&Wl[(size_t)(n0 + r) * K_DIM + (k0 + c)];
      brReg[i] = *(const float4*)&Wr[(size_t)(n0 + r) * K_DIM + (k0 + c)];
    }
  };

  auto commitLds = [&]() {
#pragma unroll
    for (int i = 0; i < 2; ++i) {
      int idx = i * 128 + tid;
      int r = idx >> 4, c = (idx & 15) * 4;
      v4h h = {(_Float16)aReg[i].x, (_Float16)aReg[i].y,
               (_Float16)aReg[i].z, (_Float16)aReg[i].w};
      *(v4h*)&sA[r][c] = h;
    }
#pragma unroll
    for (int i = 0; i < 8; ++i) {
      int idx = i * 128 + tid;
      int r = idx >> 4, c = (idx & 15) * 4;
      v4h hl = {(_Float16)blReg[i].x, (_Float16)blReg[i].y,
                (_Float16)blReg[i].z, (_Float16)blReg[i].w};
      *(v4h*)&sBl[r][c] = hl;
      v4h hr = {(_Float16)brReg[i].x, (_Float16)brReg[i].y,
                (_Float16)brReg[i].z, (_Float16)brReg[i].w};
      *(v4h*)&sBr[r][c] = hr;
    }
  };

  v8f accL0 = {}, accL1 = {}, accR0 = {}, accR1 = {};
  loadRegs(0);

  for (int k0 = 0; k0 < K_DIM; k0 += 64) {
    __syncthreads();  // previous iteration's frag reads done
    commitLds();
    __syncthreads();

    // Prefetch next tile while the 4 WMMAs run on this one.
    if (k0 + 64 < K_DIM) loadRegs(k0 + 64);

    union { v16h v; v8h h[2]; } a0, a1, bl0, bl1, br0, br1;
    a0.h[0]  = *(const v8h*)&sA [mrow][hi * 16];
    a0.h[1]  = *(const v8h*)&sA [mrow][hi * 16 + 8];
    a1.h[0]  = *(const v8h*)&sA [mrow][32 + hi * 16];
    a1.h[1]  = *(const v8h*)&sA [mrow][32 + hi * 16 + 8];
    bl0.h[0] = *(const v8h*)&sBl[ncol][hi * 16];
    bl0.h[1] = *(const v8h*)&sBl[ncol][hi * 16 + 8];
    bl1.h[0] = *(const v8h*)&sBl[ncol][32 + hi * 16];
    bl1.h[1] = *(const v8h*)&sBl[ncol][32 + hi * 16 + 8];
    br0.h[0] = *(const v8h*)&sBr[ncol][hi * 16];
    br0.h[1] = *(const v8h*)&sBr[ncol][hi * 16 + 8];
    br1.h[0] = *(const v8h*)&sBr[ncol][32 + hi * 16];
    br1.h[1] = *(const v8h*)&sBr[ncol][32 + hi * 16 + 8];

    accL0 = __builtin_amdgcn_wmma_f32_16x16x32_f16(
        false, a0.v, false, bl0.v, (short)0, accL0, false, false);
    accR0 = __builtin_amdgcn_wmma_f32_16x16x32_f16(
        false, a0.v, false, br0.v, (short)0, accR0, false, false);
    accL1 = __builtin_amdgcn_wmma_f32_16x16x32_f16(
        false, a1.v, false, bl1.v, (short)0, accL1, false, false);
    accR1 = __builtin_amdgcn_wmma_f32_16x16x32_f16(
        false, a1.v, false, br1.v, (short)0, accR1, false, false);
  }

  const int nout = n0 + wave * 16 + (lane & 15);
#pragma unroll
  for (int r = 0; r < 8; ++r) {
    int m = m0 + r + 8 * hi;
    Cl[(size_t)m * N + nout] = accL0[r] + accL1[r];
    Cr[(size_t)m * N + nout] = accR0[r] + accR1[r];
  }
}

// ---------------------------------------------------------------------------
// Helpers: zero, degree count, edge scatter (post-projection), finalize.
// ---------------------------------------------------------------------------
__global__ void zero_kernel(float* __restrict__ p, int n) {
  int i = blockIdx.x * blockDim.x + threadIdx.x;
  if (i < n) p[i] = 0.0f;
}

__global__ void count_kernel(const int* __restrict__ dst, float* __restrict__ cnt, int E) {
  int e = blockIdx.x * blockDim.x + threadIdx.x;
  if (e < E) unsafeAtomicAdd(&cnt[dst[e]], 1.0f);
}

// One block per edge, one thread per feature channel (D = blockDim.x).
__global__ void scatter_kernel(const float* __restrict__ feat,
                               const int* __restrict__ src,
                               const int* __restrict__ dst,
                               float* __restrict__ acc, int D) {
  int e = blockIdx.x;
  int s = src[e];
  int d = dst[e];
  int f = threadIdx.x;
  unsafeAtomicAdd(&acc[(size_t)d * D + f], feat[(size_t)s * D + f]);
}

// out = [relu](acc/max(cnt,1) + selfp + bias)
__global__ void finalize_kernel(const float* __restrict__ acc,
                                const float* __restrict__ selfp,
                                const float* __restrict__ bias,
                                const float* __restrict__ cnt,
                                float* __restrict__ out,
                                int total, int D, int do_relu) {
  int i = blockIdx.x * blockDim.x + threadIdx.x;
  if (i >= total) return;
  int n = i / D;
  int f = i - n * D;
  float v = acc[i] / fmaxf(cnt[n], 1.0f) + selfp[i] + bias[f];
  out[i] = do_relu ? fmaxf(v, 0.0f) : v;
}

// ---------------------------------------------------------------------------
// Launch: dual-project (WMMA) -> scatter-mean -> combine, twice.
// Mean-aggregation commutes with the linear projection, so we project first
// and aggregate in the low-dim space (4x less scatter traffic for layer 1);
// fusing the l/r projections reads X once per layer.
// ---------------------------------------------------------------------------
extern "C" void kernel_launch(void* const* d_in, const int* in_sizes, int n_in,
                              void* d_out, int out_size, void* d_ws, size_t ws_size,
                              hipStream_t stream) {
  const float* x   = (const float*)d_in[0];
  const int*   ei  = (const int*)d_in[1];
  const float* W1l = (const float*)d_in[2];
  const float* W1r = (const float*)d_in[3];
  const float* b1  = (const float*)d_in[4];
  const float* W2l = (const float*)d_in[5];
  const float* W2r = (const float*)d_in[6];
  const float* b2  = (const float*)d_in[7];
  float* out = (float*)d_out;

  const int N = 10000, E = 160000, HID = 128, OUTD = 64;
  const int* srcv = ei;
  const int* dstv = ei + E;

  float* ws = (float*)d_ws;
  size_t o = 0;
  float* xl   = ws + o; o += (size_t)N * HID;
  float* xr   = ws + o; o += (size_t)N * HID;
  float* acc1 = ws + o; o += (size_t)N * HID;
  float* h    = ws + o; o += (size_t)N * HID;
  float* hl   = ws + o; o += (size_t)N * OUTD;
  float* hr   = ws + o; o += (size_t)N * OUTD;
  float* acc2 = ws + o; o += (size_t)N * OUTD;
  float* cnt  = ws + o; o += (size_t)N;

  // Zero accumulators + degree counts (required every call: graph replay).
  zero_kernel<<<(N * HID + 255) / 256, 256, 0, stream>>>(acc1, N * HID);
  zero_kernel<<<(N * OUTD + 255) / 256, 256, 0, stream>>>(acc2, N * OUTD);
  zero_kernel<<<(N + 255) / 256, 256, 0, stream>>>(cnt, N);
  count_kernel<<<(E + 255) / 256, 256, 0, stream>>>(dstv, cnt, E);

  // ----- Layer 1: dual-project, aggregate in 128-dim space, combine + ReLU.
  sage_gemm_dual<512><<<dim3(N / 16, HID / 64), 128, 0, stream>>>(
      x, W1l, W1r, xl, xr, N, HID);
  scatter_kernel<<<E, HID, 0, stream>>>(xl, srcv, dstv, acc1, HID);
  finalize_kernel<<<(N * HID + 255) / 256, 256, 0, stream>>>(
      acc1, xr, b1, cnt, h, N * HID, HID, /*relu=*/1);

  // ----- Layer 2: same pattern in 64-dim space, no ReLU.
  sage_gemm_dual<128><<<dim3(N / 16, OUTD / 64), 128, 0, stream>>>(
      h, W2l, W2r, hl, hr, N, OUTD);
  scatter_kernel<<<E, OUTD, 0, stream>>>(hl, srcv, dstv, acc2, OUTD);
  finalize_kernel<<<(N * OUTD + 255) / 256, 256, 0, stream>>>(
      acc2, hr, b2, cnt, out, N * OUTD, OUTD, /*relu=*/0);
}
